// ExllamaLinear_52475910423189
// MI455X (gfx1250) — matrix operand order repaired and checked by reading, not compiled, not run on hardware
//
#include <hip/hip_runtime.h>

typedef _Float16 v8h  __attribute__((ext_vector_type(8)));
typedef _Float16 v16h __attribute__((ext_vector_type(16)));
typedef _Float16 h2   __attribute__((ext_vector_type(2)));
typedef float    v8f  __attribute__((ext_vector_type(8)));
typedef unsigned int u32x4 __attribute__((ext_vector_type(4)));
typedef int          i32x4 __attribute__((ext_vector_type(4)));
typedef int          i32x8 __attribute__((ext_vector_type(8)));

#define K_DIM 4096
#define N_DIM 11008
#define GS    128
#define BM    128
#define BN    128
#define BK    128            // == GROUP_SIZE: one scale/zero per k-tile
#define SA    (BK + 8)       // LDS row stride (fp16): 64 DWORDs data + 4 DWORDs pad
#define SB    (BK + 8)
#define NT    (K_DIM / BK)   // 32 k-tiles

// ---- Tensor Data Mover: load one BM x BK fp16 tile (row-major, stride K_DIM)
// into LDS with hardware row padding matching SA = BK+8 fp16 elems.
// D# layout per CDNA5 ISA §8 (group0 128b, group1 256b, groups 2/3 zero => 2D).
__device__ __forceinline__ void tdm_load_tile(const _Float16* gsrc,
                                              unsigned lds_addr, int dim1)
{
    const unsigned long long ga = (unsigned long long)(size_t)gsrc;

    u32x4 g0;
    g0[0] = 1u;                                         // count=1 user descriptor
    g0[1] = lds_addr;                                   // lds_addr [63:32]
    g0[2] = (unsigned)(ga & 0xFFFFFFFFu);               // global_addr lo
    g0[3] = (unsigned)((ga >> 32) & 0x01FFFFFFu)        // global_addr [56:32]
          | (2u << 30);                                 // type=2 ("image")

    i32x8 g1;
    g1[0] = (1 << 16)                                   // data_size = 2 bytes
          | (1 << 20)                                   // pad_enable
          | (5 << 22)                                   // pad_interval: 64 DWORDs
          | (3 << 25);                                  // pad_amount:   4 DWORDs
    g1[1] = (int)((K_DIM & 0xFFFF) << 16);              // tensor_dim0 lo16
    g1[2] = (int)((K_DIM >> 16) | ((dim1 & 0xFFFF) << 16)); // dim0 hi | dim1 lo
    g1[3] = (int)(((unsigned)dim1 >> 16) | (BM << 16)); // dim1 hi | tile_dim0=128
    g1[4] = BM;                                         // tile_dim1=128, tile_dim2=0
    g1[5] = K_DIM;                                      // tensor_dim0_stride lo32
    g1[6] = 0;                                          // stride0 hi16 | stride1 lo16
    g1[7] = 0;

    i32x4 g2 = {0, 0, 0, 0};
    i32x4 g3 = {0, 0, 0, 0};

#if __clang_major__ >= 23
    i32x8 gz = {0, 0, 0, 0, 0, 0, 0, 0};
    __builtin_amdgcn_tensor_load_to_lds(g0, g1, g2, g3, gz, 0);
#else
    __builtin_amdgcn_tensor_load_to_lds(g0, g1, g2, g3, 0);
#endif
}

// ---- int4 -> fp16 magic dequant: (q | 0x6400) as fp16 == 1024 + q.
// w = pk_fma(h, {s,s}, {zf,zf}) with zf = -(1025+z)*s  =>  (q - (z+1)) * s
__device__ __forceinline__ v8h dequant8(unsigned q32, h2 s2, h2 zf2)
{
    const unsigned p01 = (q32 & 0xFu)          | ((q32 << 12) & 0x000F0000u);
    const unsigned p23 = ((q32 >> 8) & 0xFu)   | ((q32 << 4)  & 0x000F0000u);
    const unsigned p45 = ((q32 >> 16) & 0xFu)  | ((q32 >> 4)  & 0x000F0000u);
    const unsigned p67 = ((q32 >> 24) & 0xFu)  | ((q32 >> 12) & 0x000F0000u);
    const h2 w01 = __builtin_elementwise_fma(
        __builtin_bit_cast(h2, p01 | 0x64006400u), s2, zf2);
    const h2 w23 = __builtin_elementwise_fma(
        __builtin_bit_cast(h2, p23 | 0x64006400u), s2, zf2);
    const h2 w45 = __builtin_elementwise_fma(
        __builtin_bit_cast(h2, p45 | 0x64006400u), s2, zf2);
    const h2 w67 = __builtin_elementwise_fma(
        __builtin_bit_cast(h2, p67 | 0x64006400u), s2, zf2);
    v8h w;
    w[0] = w01[0]; w[1] = w01[1];
    w[2] = w23[0]; w[3] = w23[1];
    w[4] = w45[0]; w[5] = w45[1];
    w[6] = w67[0]; w[7] = w67[1];
    return w;
}

__global__ __launch_bounds__(256)
void exllama_w4a16_wmma(const _Float16* __restrict__ x,
                        const int*      __restrict__ qweight,
                        const int*      __restrict__ qzeros,
                        const _Float16* __restrict__ scales,
                        const _Float16* __restrict__ bias,
                        _Float16*       __restrict__ out,
                        int m_total)
{
    __shared__ _Float16 As[2][BM * SA];   // x tile ping-pong (TDM destination)
    __shared__ _Float16 Bs[BN * SB];      // dequantized W tile, N-major [n][k]

    const int t    = threadIdx.x;
    const int lane = t & 31;
    const int wave = t >> 5;             // 8 waves
    const int wm   = wave >> 2;          // 0..1  -> 64 M-rows each
    const int wn   = wave & 3;           // 0..3  -> 32 N-cols each
    const int hi   = lane >> 4;          // lane half (K-chunk select)
    const int l15  = lane & 15;

    const int n0 = blockIdx.x * BN;
    const int m0 = blockIdx.y * BM;

    // B-dequant mapping: 2 threads per column, 8 packed int32 rows each
    const int nl     = t & 127;          // local column
    const int khalf  = t >> 7;           // packed rows [0..7] or [8..15]
    const int ng     = n0 + nl;          // global column
    const int zshift = (ng & 7) * 4;

    const unsigned ldsA0 = (unsigned)(size_t)(&As[0][0]);
    const unsigned ldsA1 = (unsigned)(size_t)(&As[1][0]);

    v8f acc[4][2] = {};                  // 4 M-tiles x 2 N-tiles of 16x16 f32

    // Prime the pipeline: TDM fetch of k-tile 0 into As[0]
    if (wave == 0)
        tdm_load_tile(x + (size_t)m0 * K_DIM, ldsA0, m_total);

    for (int kt = 0; kt < NT; ++kt) {
        const int k0  = kt * BK;
        const int buf = kt & 1;
        const _Float16* __restrict__ Ab = &As[buf][0];

        // ---- dequantize B tile: packed int4 -> fp16, N-major in LDS ----
        {
            const float sf = (float)scales[(size_t)kt * N_DIM + ng];
            const int   zq = (qzeros[(size_t)kt * (N_DIM / 8) + (ng >> 3)]
                              >> zshift) & 15;
            const float zff = -(float)(1025 + zq) * sf;   // -(1024 + z + 1)*s
            const _Float16 sh = (_Float16)sf, zh = (_Float16)zff;
            const h2 s2  = {sh, sh};
            const h2 zf2 = {zh, zh};
            #pragma unroll
            for (int j = 0; j < 8; ++j) {
                const int pr = khalf * 8 + j;             // packed k-row in tile
                const unsigned q32 =
                    (unsigned)qweight[(size_t)(k0 / 8 + pr) * N_DIM + ng];
                *(v8h*)(&Bs[nl * SB + pr * 8]) = dequant8(q32, s2, zf2);
            }
        }

        // prefetch next k-tile's packed weights while we compute
        if (kt + 1 < NT) {
            __builtin_prefetch(
                qweight + (size_t)(k0 / 8 + 16 + khalf * 8) * N_DIM + ng, 0, 1);
        }

        // ---- software pipeline: launch TDM for kt+1, wait for kt ----
        if (wave == 0) {
            if (kt + 1 < NT) {
                tdm_load_tile(x + (size_t)m0 * K_DIM + (kt + 1) * BK,
                              buf ? ldsA0 : ldsA1, m_total);
                __builtin_amdgcn_s_wait_tensorcnt(1);   // kt's tile landed
            } else {
                __builtin_amdgcn_s_wait_tensorcnt(0);
            }
        }
        __syncthreads();

        // ---- 4 K-steps of 32, 8 WMMAs each per wave ----
        #pragma unroll
        for (int kk = 0; kk < BK; kk += 32) {
            // A fragments (ISA 16-bit A 16x32 layout):
            //  lane half 0: K = kk+[0..7], kk+16+[0..7]
            //  lane half 1: K = kk+8+[0..7], kk+24+[0..7]
            v16h afrag[4];
            #pragma unroll
            for (int mt = 0; mt < 4; ++mt) {
                const _Float16* ap =
                    &Ab[(wm * 64 + mt * 16 + l15) * SA + kk + hi * 8];
                v8h lo = *(const v8h*)(ap);
                v8h hv = *(const v8h*)(ap + 16);
                afrag[mt] = __builtin_shufflevector(
                    lo, hv, 0, 1, 2, 3, 4, 5, 6, 7,
                            8, 9, 10, 11, 12, 13, 14, 15);
            }
            // B fragments: lane = column, half 0: K 0-7,16-23 / half 1: K 8-15,24-31
            v16h bfrag[2];
            #pragma unroll
            for (int nt = 0; nt < 2; ++nt) {
                const _Float16* bp =
                    &Bs[(wn * 32 + nt * 16 + l15) * SB + kk + hi * 8];
                v8h lo = *(const v8h*)(bp);
                v8h hv = *(const v8h*)(bp + 16);
                bfrag[nt] = __builtin_shufflevector(
                    lo, hv, 0, 1, 2, 3, 4, 5, 6, 7,
                            8, 9, 10, 11, 12, 13, 14, 15);
            }
            #pragma unroll
            for (int mt = 0; mt < 4; ++mt)
                #pragma unroll
                for (int nt = 0; nt < 2; ++nt)
                    acc[mt][nt] = __builtin_amdgcn_wmma_f32_16x16x32_f16(
                        false, afrag[mt], false, bfrag[nt],
                        (short)0, acc[mt][nt], false, false);
        }

        __syncthreads();
    }

    // ---- epilogue: bias add in f32, convert to fp16, store ----
    // C layout: VGPR r, lane l -> M = r + 8*(l>=16), N = l&15
    #pragma unroll
    for (int mt = 0; mt < 4; ++mt) {
        #pragma unroll
        for (int nt = 0; nt < 2; ++nt) {
            const int n  = n0 + wn * 32 + nt * 16 + l15;
            const float b = (float)bias[n];
            #pragma unroll
            for (int r = 0; r < 8; ++r) {
                const int m = m0 + wm * 64 + mt * 16 + hi * 8 + r;
                out[(size_t)m * N_DIM + n] = (_Float16)(acc[mt][nt][r] + b);
            }
        }
    }
}

extern "C" void kernel_launch(void* const* d_in, const int* in_sizes, int n_in,
                              void* d_out, int out_size, void* d_ws, size_t ws_size,
                              hipStream_t stream)
{
    const _Float16* x       = (const _Float16*)d_in[0];
    const int*      qweight = (const int*)d_in[1];
    const int*      qzeros  = (const int*)d_in[2];
    const _Float16* scales  = (const _Float16*)d_in[3];
    const _Float16* bias    = (const _Float16*)d_in[4];
    _Float16*       out     = (_Float16*)d_out;

    const int m_total = in_sizes[0] / K_DIM;     // 4*2048 = 8192
    dim3 grid(N_DIM / BN, m_total / BM);         // (86, 64)
    dim3 block(256);
    exllama_w4a16_wmma<<<grid, block, 0, stream>>>(x, qweight, qzeros,
                                                   scales, bias, out, m_total);
}
